// RNN_18193481466133
// MI455X (gfx1250) — compile-verified, matrix operand order checked by reading, not run on hardware
//
#include <hip/hip_runtime.h>
#include <hip/hip_bf16.h>

// ---------------------------------------------------------------------------
// 2-layer tanh RNN + linear decoder for MI455X (gfx1250, wave32, WMMA).
//   1) f32 -> bf16 conversions, bias folding
//   2) xw = X * W_ih^T + b   (WMMA GEMM, 128x64 WG tile, A-frag reuse x4)
//   3) persistent 8-WG recurrence: h_t = tanh(xw_t + h_{t-1} W_hh^T)
//      W_hh slice + full h in LDS; h exchanged through L2 each step with a
//      device-scope atomic barrier; global->LDS copies use the gfx1250
//      async-to-LDS DMA path (ASYNCcnt) when available.
//   4) decoder GEMM into d_out.
// ---------------------------------------------------------------------------

typedef __attribute__((ext_vector_type(16))) __bf16 v16bf;
typedef __attribute__((ext_vector_type(8)))  float  v8f;
typedef __attribute__((ext_vector_type(4)))  int    v4i;

constexpr int Bc = 32, Sc = 2048, Ic = 256, Hc = 512, Oc = 256;
constexpr int NWG = 8;                   // workgroups in persistent recurrence
constexpr int COLS_PER_WG = Hc / NWG;    // 64 output columns per WG

#ifndef __has_builtin
#define __has_builtin(x) 0
#endif
#if __has_builtin(__builtin_amdgcn_global_load_async_to_lds_b128)
#define HAVE_ASYNC_LDS 1
#else
#define HAVE_ASYNC_LDS 0
#endif

typedef __attribute__((address_space(1))) v4i gv4i;   // global int4
typedef __attribute__((address_space(3))) v4i lv4i;   // LDS int4

// 16-byte global -> LDS copy; async DMA (ASYNCcnt) on gfx1250 toolchains that
// expose it, else VGPR round-trip.
__device__ __forceinline__ void copy_b128_to_lds(const void* g, void* l) {
#if HAVE_ASYNC_LDS
  __builtin_amdgcn_global_load_async_to_lds_b128((gv4i*)g, (lv4i*)l, 0, 0);
#else
  *(uint4*)l = *(const uint4*)g;
#endif
}

__device__ __forceinline__ void async_lds_wait0() {
#if HAVE_ASYNC_LDS
#if __has_builtin(__builtin_amdgcn_s_wait_asynccnt)
  __builtin_amdgcn_s_wait_asynccnt(0);
#else
  asm volatile("s_wait_asynccnt 0" ::: "memory");
#endif
#endif
}

union Frag16 { uint4 q[2]; v16bf v; };

// Load a 16-element bf16 fragment row: K = [kb..kb+7] and [kb+16..kb+23]
// (per CDNA5 ISA 16-bit A/B 16x32 VGPR layout; lane half selects kb or kb+8).
__device__ __forceinline__ v16bf load_frag(const __bf16* p) {
  Frag16 u;
  const uint4* q = reinterpret_cast<const uint4*>(p);
  u.q[0] = q[0];   // K = kb .. kb+7
  u.q[1] = q[2];   // K = kb+16 .. kb+23
  return u.v;
}

__global__ void cvt_f32_bf16(const float* __restrict__ src,
                             __bf16* __restrict__ dst, int n) {
  int i = blockIdx.x * blockDim.x + threadIdx.x;
  if (i < n) dst[i] = (__bf16)src[i];
}

__global__ void bias_sum(const float* __restrict__ a, const float* __restrict__ b,
                         float* __restrict__ o, int n) {
  int i = blockIdx.x * blockDim.x + threadIdx.x;
  if (i < n) o[i] = a[i] + b[i];
}

// C[m,n] = sum_k A[m,k] * W[n,k] (+ bias[n]).  A:[M,K] bf16 row-major,
// W:[N,K] bf16 row-major, C:[M,N] f32.
// Block = 8 waves; WG tile 128(M) x 64(N); each wave owns one 16-row M tile
// and all four 16-col N tiles, reusing its A fragment 4x per k-step.
__global__ void wmma_gemm_bias(const __bf16* __restrict__ A,
                               const __bf16* __restrict__ W,
                               const float* __restrict__ bias,
                               float* __restrict__ C, int K) {
  const int N     = gridDim.x * 64;
  const int lane  = threadIdx.x & 31;
  const int wave  = threadIdx.x >> 5;
  const int m0    = blockIdx.y * 128 + wave * 16;
  const int n0    = blockIdx.x * 64;
  const int lrow  = lane & 15;
  const int lhalf = lane >> 4;

  const __bf16* arow = A + (size_t)(m0 + lrow) * K + (lhalf << 3);
  const __bf16* wrow = W + (size_t)(n0 + lrow) * K + (lhalf << 3);

  v8f acc[4] = {{}, {}, {}, {}};
#pragma unroll 2
  for (int k0 = 0; k0 < K; k0 += 32) {
    __builtin_prefetch(arow + k0 + 128, 0, 3);   // global_prefetch_b8, near RT
    v16bf a  = load_frag(arow + k0);
    v16bf b0 = load_frag(wrow + (size_t)0 * 16 * K + k0);
    v16bf b1 = load_frag(wrow + (size_t)1 * 16 * K + k0);
    v16bf b2 = load_frag(wrow + (size_t)2 * 16 * K + k0);
    v16bf b3 = load_frag(wrow + (size_t)3 * 16 * K + k0);
    acc[0] = __builtin_amdgcn_wmma_f32_16x16x32_bf16(false, a, false, b0,
                                                     (short)0, acc[0], false, false);
    acc[1] = __builtin_amdgcn_wmma_f32_16x16x32_bf16(false, a, false, b1,
                                                     (short)0, acc[1], false, false);
    acc[2] = __builtin_amdgcn_wmma_f32_16x16x32_bf16(false, a, false, b2,
                                                     (short)0, acc[2], false, false);
    acc[3] = __builtin_amdgcn_wmma_f32_16x16x32_bf16(false, a, false, b3,
                                                     (short)0, acc[3], false, false);
  }
#pragma unroll
  for (int nt = 0; nt < 4; ++nt) {
    const int   col = n0 + nt * 16 + lrow;
    const float bv  = bias ? bias[col] : 0.0f;
#pragma unroll
    for (int r = 0; r < 8; ++r) {
      const int m = m0 + r + lhalf * 8;          // C/D layout: M = vgpr + 8*half
      C[(size_t)m * N + col] = acc[nt][r] + bv;
    }
  }
}

// Persistent recurrence over all S steps of one layer.
// Grid = NWG blocks x 256 threads. Block wg owns output columns
// [wg*64, wg*64+64). Dynamic LDS: W slice (64x512 bf16) + full h (32x512 bf16).
__global__ void rnn_recurrence(const float* __restrict__ xw,     // [B,S,H] f32
                               const __bf16* __restrict__ Whh,   // [H,H] bf16
                               const float* __restrict__ h0,     // [B,H] f32
                               __bf16* __restrict__ y_bf,        // [B,S,H] bf16 out
                               float* __restrict__ hT,           // [B,H] f32 out
                               __bf16* __restrict__ h_exch,      // [2,B,H] bf16
                               int* __restrict__ counter) {
  extern __shared__ char smem_raw[];
  __bf16* sW = reinterpret_cast<__bf16*>(smem_raw);                    // 64*512
  __bf16* sH = reinterpret_cast<__bf16*>(smem_raw) + COLS_PER_WG * Hc; // 32*512

  const int tid   = threadIdx.x;
  const int nbase = blockIdx.x * COLS_PER_WG;

  { // stage this WG's 64-row slice of W_hh into LDS (async DMA when available)
    const char* src = (const char*)(Whh + (size_t)nbase * Hc);
    char* dst = (char*)sW;
    for (int i = tid; i < COLS_PER_WG * Hc * 2 / 16; i += blockDim.x)
      copy_b128_to_lds(src + (size_t)i * 16, dst + (size_t)i * 16);
    async_lds_wait0();
  }
  for (int i = tid; i < Bc * Hc; i += blockDim.x) sH[i] = (__bf16)h0[i];
  __syncthreads();

  const int lane  = tid & 31;
  const int wave  = tid >> 5;
  const int m0    = (wave >> 2) * 16;     // batch-row tile (0 or 16)
  const int n0l   = (wave & 3) * 16;      // local col tile
  const int lrow  = lane & 15;
  const int lhalf = lane >> 4;
  const __bf16* wrow  = sW + (size_t)(n0l + lrow) * Hc + (lhalf << 3);
  const __bf16* hrow  = sH + (size_t)(m0 + lrow) * Hc + (lhalf << 3);
  const int     col_g = nbase + n0l + lrow;

  for (int s = 0; s < Sc; ++s) {
    v8f acc = {};
#pragma unroll
    for (int k0 = 0; k0 < Hc; k0 += 32) {
      v16bf a = load_frag(hrow + k0);
      v16bf b = load_frag(wrow + k0);
      acc = __builtin_amdgcn_wmma_f32_16x16x32_bf16(false, a, false, b,
                                                    (short)0, acc, false, false);
    }
    __bf16* hx = h_exch + (size_t)(s & 1) * Bc * Hc;  // double buffer
#pragma unroll
    for (int r = 0; r < 8; ++r) {
      const int    b    = m0 + r + lhalf * 8;
      const size_t xoff = ((size_t)b * Sc + s) * Hc + col_g;
      float  v  = tanhf(acc[r] + xw[xoff]);
      __bf16 vb = (__bf16)v;
      y_bf[xoff]           = vb;
      hx[b * Hc + col_g]   = vb;
      if (s == Sc - 1) hT[b * Hc + col_g] = v;
      if (s + 1 < Sc)   // warm L2/near cache for next step's xw row
        __builtin_prefetch(xw + xoff + Hc, 0, 3);
    }
    // ---- device-scope barrier across the NWG persistent workgroups ----
    __threadfence();
    __syncthreads();
    if (tid == 0) {
      atomicAdd(counter, 1);
      const int target = NWG * (s + 1);
      while (__hip_atomic_load(counter, __ATOMIC_ACQUIRE,
                               __HIP_MEMORY_SCOPE_AGENT) < target)
        __builtin_amdgcn_s_sleep(1);
    }
    __syncthreads();
    __threadfence();
    { // async-gather full h_t (all 512 cols) into LDS for the next step
      const char* src = (const char*)hx;
      char* dst = (char*)sH;
      for (int i = tid; i < Bc * Hc * 2 / 16; i += blockDim.x)
        copy_b128_to_lds(src + (size_t)i * 16, dst + (size_t)i * 16);
      async_lds_wait0();
    }
    __syncthreads();
  }
}

extern "C" void kernel_launch(void* const* d_in, const int* in_sizes, int n_in,
                              void* d_out, int out_size, void* d_ws, size_t ws_size,
                              hipStream_t stream) {
  const float* x    = (const float*)d_in[0];   // [B,S,I]
  const float* h0s  = (const float*)d_in[1];   // [L,B,H]
  const float* wih0 = (const float*)d_in[2];   // [H,I]
  const float* whh0 = (const float*)d_in[3];   // [H,H]
  const float* bih0 = (const float*)d_in[4];
  const float* bhh0 = (const float*)d_in[5];
  const float* wih1 = (const float*)d_in[6];   // [H,H]
  const float* whh1 = (const float*)d_in[7];   // [H,H]
  const float* bih1 = (const float*)d_in[8];
  const float* bhh1 = (const float*)d_in[9];
  const float* decw = (const float*)d_in[10];  // [O,H]
  float* out = (float*)d_out;                  // [B,S,O] ++ [L,B,H]

  char* basep = (char*)d_ws;
  size_t off = 0;
  auto alloc = [&](size_t bytes) -> void* {
    void* p = basep + off;
    off += (bytes + 255) & ~(size_t)255;
    return p;
  };
  int*    counters = (int*)   alloc(256);
  __bf16* x_bf     = (__bf16*)alloc((size_t)Bc * Sc * Ic * 2);
  __bf16* y_bf     = (__bf16*)alloc((size_t)Bc * Sc * Hc * 2);
  float*  xw       = (float*) alloc((size_t)Bc * Sc * Hc * 4);
  __bf16* wih0_bf  = (__bf16*)alloc((size_t)Hc * Ic * 2);
  __bf16* whh0_bf  = (__bf16*)alloc((size_t)Hc * Hc * 2);
  __bf16* wih1_bf  = (__bf16*)alloc((size_t)Hc * Hc * 2);
  __bf16* whh1_bf  = (__bf16*)alloc((size_t)Hc * Hc * 2);
  __bf16* dec_bf   = (__bf16*)alloc((size_t)Oc * Hc * 2);
  float*  bias0    = (float*) alloc(Hc * 4);
  float*  bias1    = (float*) alloc(Hc * 4);
  __bf16* h_exch   = (__bf16*)alloc((size_t)2 * Bc * Hc * 2);

  (void)hipMemsetAsync(counters, 0, 256, stream);   // barrier counters = 0

  auto cvt = [&](const float* s, __bf16* d, int n) {
    cvt_f32_bf16<<<(n + 255) / 256, 256, 0, stream>>>(s, d, n);
  };
  cvt(x,    x_bf,    Bc * Sc * Ic);
  cvt(wih0, wih0_bf, Hc * Ic);
  cvt(whh0, whh0_bf, Hc * Hc);
  cvt(wih1, wih1_bf, Hc * Hc);
  cvt(whh1, whh1_bf, Hc * Hc);
  cvt(decw, dec_bf,  Oc * Hc);
  bias_sum<<<(Hc + 255) / 256, 256, 0, stream>>>(bih0, bhh0, bias0, Hc);
  bias_sum<<<(Hc + 255) / 256, 256, 0, stream>>>(bih1, bhh1, bias1, Hc);

  const dim3 blk(256);
  const dim3 gH(Hc / 64, (Bc * Sc) / 128);  // N=512 projection GEMMs
  const dim3 gO(Oc / 64, (Bc * Sc) / 128);  // N=256 decoder GEMM
  const size_t smem = (size_t)(COLS_PER_WG * Hc + Bc * Hc) * 2;  // 96 KB

  float* hid_out = out + (size_t)Bc * Sc * Oc;

  // Layer 0
  wmma_gemm_bias<<<gH, blk, 0, stream>>>(x_bf, wih0_bf, bias0, xw, Ic);
  rnn_recurrence<<<NWG, blk, smem, stream>>>(xw, whh0_bf, h0s, y_bf,
                                             hid_out, h_exch, counters + 0);
  // Layer 1
  wmma_gemm_bias<<<gH, blk, 0, stream>>>(y_bf, wih1_bf, bias1, xw, Hc);
  rnn_recurrence<<<NWG, blk, smem, stream>>>(xw, whh1_bf, h0s + Bc * Hc, y_bf,
                                             hid_out + Bc * Hc, h_exch,
                                             counters + 1);
  // Decoder
  wmma_gemm_bias<<<gO, blk, 0, stream>>>(y_bf, dec_bf, nullptr, out, Hc);
}
